// LinformerSelfAttention_25649544692321
// MI455X (gfx1250) — matrix operand order repaired and checked by reading, not compiled, not run on hardware
//
#include <hip/hip_runtime.h>
#include <hip/hip_bf16.h>

// ---------------------------------------------------------------------------
// Linformer self-attention for MI455X (gfx1250), wave32, all GEMMs via
// v_wmma_f32_16x16x32_f16, LDS fragments via ds_load_b128, attention K/V/Q
// staging via the Tensor Data Mover (tensor_load_to_lds + s_wait_tensorcnt).
// ---------------------------------------------------------------------------

constexpr int cB  = 8;
constexpr int cS  = 4096;
constexpr int cD  = 1024;
constexpr int cH  = 16;
constexpr int cK  = 256;
constexpr int cDH = 64;

typedef __attribute__((ext_vector_type(16))) _Float16 v16h;
typedef __attribute__((ext_vector_type(8)))  _Float16 v8h;
typedef __attribute__((ext_vector_type(8)))  float    v8f;
typedef __attribute__((ext_vector_type(4)))  unsigned int u32x4;
typedef __attribute__((ext_vector_type(8)))  int      i32x8;
typedef __attribute__((ext_vector_type(4)))  int      i32x4;

#if __has_builtin(__builtin_amdgcn_tensor_load_to_lds) && \
    __has_builtin(__builtin_amdgcn_s_wait_tensorcnt)
#define HAVE_TDM 1
#else
#define HAVE_TDM 0
#endif

static __device__ __forceinline__ int lane_id() { return threadIdx.x & 31; }

#if HAVE_TDM
// 1-D TDM copy: n8 contiguous 8-byte elements, global -> LDS.
// D# per CDNA5 ISA §8.3/8.4: group0 {count=1, lds_addr, global_addr, type=2},
// group1 {mask=0, data_size=3, tensor_dim0=tile_dim0=n8, dims1=1, stride=n8}.
// 6-arg builtin: (g0, g1, g2, g3, g4, cpol) — groups 2/3/4 zero (<=2D tensor).
static __device__ __forceinline__ void tdm_load_1d(const void* gsrc, void* ldst,
                                                   int n8) {
  unsigned long long ga = (unsigned long long)gsrc;
  unsigned la = (unsigned)(unsigned long long)ldst;  // flat LDS addr[31:0] = LDS offset
  u32x4 g0;
  g0[0] = 1u;                                        // count=1, user descriptor
  g0[1] = la;                                        // lds_addr (bytes)
  g0[2] = (unsigned)(ga & 0xffffffffu);              // global_addr[31:0]
  g0[3] = (unsigned)((ga >> 32) & 0x1ffffffu) | (2u << 30);  // addr[56:32], type=2
  i32x8 g1;
  g1[0] = 3 << 16;                                   // data_size=3 (8B), wg_mask=0
  g1[1] = (int)((unsigned)(n8 & 0xffff) << 16);      // tensor_dim0[15:0]
  g1[2] = (int)(((unsigned)n8 >> 16) & 0xffffu) | (1 << 16);  // td0 hi, tensor_dim1=1
  g1[3] = (int)((unsigned)(n8 & 0xffff) << 16);      // tile_dim0
  g1[4] = 1;                                         // tile_dim1=1, tile_dim2=0
  g1[5] = n8;                                        // tensor_dim0_stride lo
  g1[6] = 0;
  g1[7] = 0;
  i32x4 z4 = {0, 0, 0, 0};
  i32x8 z8 = {0, 0, 0, 0, 0, 0, 0, 0};
  __builtin_amdgcn_tensor_load_to_lds(g0, g1, z4, z4, z8, 0);
}
#endif

// A-matrix fragment (16x32 f16) from row-major LDS (stride lda halves).
static __device__ __forceinline__ v16h ldsA(const _Float16* base, int lda) {
  int lane = lane_id();
  const _Float16* r = base + (lane & 15) * lda + ((lane & 16) ? 8 : 0);
  v8h lo = *(const v8h*)r;
  v8h hi = *(const v8h*)(r + 16);
  return __builtin_shufflevector(lo, hi, 0, 1, 2, 3, 4, 5, 6, 7,
                                 8, 9, 10, 11, 12, 13, 14, 15);
}

// B-matrix fragment (32x16) from n-major LDS: elem(k,n) = base[n*ldn + k].
static __device__ __forceinline__ v16h ldsB_n(const _Float16* base, int ldn) {
  int lane = lane_id();
  const _Float16* p = base + (lane & 15) * ldn + ((lane & 16) ? 8 : 0);
  v8h lo = *(const v8h*)p;
  v8h hi = *(const v8h*)(p + 16);
  return __builtin_shufflevector(lo, hi, 0, 1, 2, 3, 4, 5, 6, 7,
                                 8, 9, 10, 11, 12, 13, 14, 15);
}

static __device__ __forceinline__ v8f wmma32(v16h a, v16h b, v8f c) {
  return __builtin_amdgcn_wmma_f32_16x16x32_f16(false, a, false, b,
                                                (short)0, c, false, false);
}

// C/D 16x16 f32 store: VGPR r -> row mbase+r, col lane&15
static __device__ __forceinline__ void storeC(float* C, int ldc, v8f c) {
  int lane = lane_id();
  float* p = C + ((lane & 16) ? 8 : 0) * ldc + (lane & 15);
#pragma unroll
  for (int r = 0; r < 8; ++r) p[r * ldc] = c[r];
}

// ---------------------------------------------------------------------------
// lengths -> 1/sqrt(len) per batch
// ---------------------------------------------------------------------------
__global__ void len_kernel(const float* __restrict__ mask, float* __restrict__ invsqrt) {
  __shared__ int red[256];
  int b = blockIdx.x;
  int cnt = 0;
  for (int s = threadIdx.x; s < cS; s += 256)
    cnt += (mask[(size_t)b * cS + s] > -1.0f) ? 1 : 0;
  red[threadIdx.x] = cnt;
  __syncthreads();
  for (int o = 128; o > 0; o >>= 1) {
    if ((int)threadIdx.x < o) red[threadIdx.x] += red[threadIdx.x + o];
    __syncthreads();
  }
  if (threadIdx.x == 0) invsqrt[b] = rsqrtf((float)red[0]);
}

// ---------------------------------------------------------------------------
// csum[proj][b][k] = sum_s m[b,s] * P[s,k]   (bias carry term)
// ---------------------------------------------------------------------------
__global__ void csum_kernel(const float* __restrict__ mask,
                            const float* __restrict__ pk,
                            const float* __restrict__ pv,
                            float* __restrict__ csum) {
  int proj = blockIdx.x, b = blockIdx.y;
  const float* P = proj ? pv : pk;
  int k = threadIdx.x;  // 256 == cK
  float s = 0.f;
  for (int ss = 0; ss < cS; ++ss)
    s += (mask[(size_t)b * cS + ss] > -1.0f) ? P[(size_t)ss * cK + k] : 0.0f;
  csum[((size_t)proj * cB + b) * cK + k] = s;
}

// ---------------------------------------------------------------------------
// Xt[b][d][s] (f16) = X[b][s][d] (f32): LDS-tiled transpose.
// grid (S/64, D/64, B), 256 threads.
// ---------------------------------------------------------------------------
__global__ __launch_bounds__(256)
void transpose_x_kernel(const float* __restrict__ X, _Float16* __restrict__ Xt) {
  __shared__ _Float16 t[64][72];  // +8 pad against bank conflicts
  int b = blockIdx.z;
  int s0 = blockIdx.x * 64, d0 = blockIdx.y * 64;
  int tid = threadIdx.x;
  {
    int sl = tid >> 2, d16 = (tid & 3) * 16;
    const float* xr = X + ((size_t)b * cS + s0 + sl) * cD + d0 + d16;
#pragma unroll
    for (int e = 0; e < 16; ++e) t[sl][d16 + e] = (_Float16)xr[e];
  }
  __syncthreads();
  {
    int dl = tid >> 2, s16 = (tid & 3) * 16;
    _Float16* xo = Xt + ((size_t)b * cD + d0 + dl) * cS + s0 + s16;
#pragma unroll
    for (int e = 0; e < 16; ++e) xo[e] = t[s16 + e][dl];
  }
}

// ---------------------------------------------------------------------------
// Fused Ck/Cv: C[b] = (m ⊙ P)^T @ X[b] : (K x S)(S x D) -> (K x D), f32 out.
// X tiles staged once (from f16 Xt, n-major), shared by both projections.
// Double-buffered. grid (D/64, K/64, B), 256 threads (8 waves).
// ---------------------------------------------------------------------------
__global__ __launch_bounds__(256)
void proj_pT_x_kernel(const _Float16* __restrict__ Xt, const float* __restrict__ mask,
                      const float* __restrict__ Pk, const float* __restrict__ Pv,
                      float* __restrict__ CkOut, float* __restrict__ CvOut) {
  __shared__ __align__(16) _Float16 Ak[2][64 * 32];  // (m pk)^T tile [k][s]
  __shared__ __align__(16) _Float16 Av[2][64 * 32];  // (m pv)^T tile [k][s]
  __shared__ __align__(16) _Float16 Bt[2][64 * 32];  // Xt tile [d][s] (n-major)
  int b  = blockIdx.z;
  int k0 = blockIdx.y * 64;
  int d0 = blockIdx.x * 64;
  int tid = threadIdx.x, w = tid >> 5;
  int tm = w & 3, tn0 = (w >> 2) * 2;
  const float* mb = mask + (size_t)b * cS;
  const _Float16* Xb = Xt + (size_t)b * cD * cS;
  int sl = tid >> 3, k8 = (tid & 7) * 8;  // A staging map
  int dl = tid >> 2, s8 = (tid & 3) * 8;  // B staging map
  float pkr[8], pvr[8];
  uint4 xreg;
  auto loadRegs = [&](int sc) {
    float m = (mb[sc + sl] > -1.0f) ? 1.0f : 0.0f;
    const float* prk = Pk + (size_t)(sc + sl) * cK + k0 + k8;
    const float* prv = Pv + (size_t)(sc + sl) * cK + k0 + k8;
#pragma unroll
    for (int e = 0; e < 8; ++e) { pkr[e] = prk[e] * m; pvr[e] = prv[e] * m; }
    xreg = *(const uint4*)(Xb + (size_t)(d0 + dl) * cS + sc + s8);
  };
  auto storeRegs = [&](int p) {
#pragma unroll
    for (int e = 0; e < 8; ++e) {
      Ak[p][(k8 + e) * 32 + sl] = (_Float16)pkr[e];
      Av[p][(k8 + e) * 32 + sl] = (_Float16)pvr[e];
    }
    *(uint4*)(Bt[p] + dl * 32 + s8) = xreg;
  };
  v8f ak0 = {}, ak1 = {}, av0 = {}, av1 = {};
  loadRegs(0);
  storeRegs(0);
  const int nIt = cS / 32;
  for (int it = 0; it < nIt; ++it) {
    __syncthreads();
    int p = it & 1;
    if (it + 1 < nIt) loadRegs((it + 1) * 32);
    v16h aK = ldsA(Ak[p] + tm * 16 * 32, 32);
    v16h aV = ldsA(Av[p] + tm * 16 * 32, 32);
    v16h b0 = ldsB_n(Bt[p] + tn0 * 16 * 32, 32);
    v16h b1 = ldsB_n(Bt[p] + (tn0 + 1) * 16 * 32, 32);
    ak0 = wmma32(aK, b0, ak0);
    ak1 = wmma32(aK, b1, ak1);
    av0 = wmma32(aV, b0, av0);
    av1 = wmma32(aV, b1, av1);
    if (it + 1 < nIt) storeRegs((it + 1) & 1);
  }
  float* Ckb = CkOut + ((size_t)b * cK + k0 + tm * 16) * cD + d0;
  float* Cvb = CvOut + ((size_t)b * cK + k0 + tm * 16) * cD + d0;
  storeC(Ckb + tn0 * 16, cD, ak0);
  storeC(Ckb + (tn0 + 1) * 16, cD, ak1);
  storeC(Cvb + tn0 * 16, cD, av0);
  storeC(Cvb + (tn0 + 1) * 16, cD, av1);
}

// ---------------------------------------------------------------------------
// Y = (A @ W^T + bias*rowmul) * (const_scale * bscale[b]) -> f16.
// out layout: transposeOut==0 -> [b][h][m][dh]; ==1 -> [b][h][dh][m].
// 128x64 macro-tile, double-buffered. grid (D/64, M/128, B), 256 threads.
// ---------------------------------------------------------------------------
__global__ __launch_bounds__(256)
void gemm_xwT_kernel(const float* __restrict__ A, const float* __restrict__ W,
                     const float* __restrict__ bias, const float* __restrict__ rowmul,
                     const float* __restrict__ bscale, float const_scale,
                     _Float16* __restrict__ out, int M, int transposeOut) {
  __shared__ __align__(16) _Float16 Al[2][128 * 32];  // A tile [m][d]
  __shared__ __align__(16) _Float16 Wl[2][64 * 32];   // W tile [n][d] (n-major)
  int b  = blockIdx.z;
  int m0 = blockIdx.y * 128;
  int n0 = blockIdx.x * 64;
  int tid = threadIdx.x, lane = tid & 31, w = tid >> 5;
  const float* Ab = A + (size_t)b * M * cD;
  int ml = tid >> 1, d16 = (tid & 1) * 16;  // A staging map (128x32)
  int nl = tid >> 2, d8 = (tid & 3) * 8;    // W staging map (64x32)
  float aR[16], wR[8];
  auto loadRegs = [&](int dc) {
    const float* ar = Ab + (size_t)(m0 + ml) * cD + dc + d16;
#pragma unroll
    for (int e = 0; e < 16; ++e) aR[e] = ar[e];
    const float* wr = W + (size_t)(n0 + nl) * cD + dc + d8;
#pragma unroll
    for (int e = 0; e < 8; ++e) wR[e] = wr[e];
  };
  auto storeRegs = [&](int p) {
    v8h h0, h1, hw;
#pragma unroll
    for (int e = 0; e < 8; ++e) {
      h0[e] = (_Float16)aR[e];
      h1[e] = (_Float16)aR[8 + e];
      hw[e] = (_Float16)wR[e];
    }
    *(v8h*)(Al[p] + ml * 32 + d16) = h0;
    *(v8h*)(Al[p] + ml * 32 + d16 + 8) = h1;
    *(v8h*)(Wl[p] + nl * 32 + d8) = hw;
  };
  v8f acc[4] = {};
  loadRegs(0);
  storeRegs(0);
  const int nIt = cD / 32;
  for (int it = 0; it < nIt; ++it) {
    __syncthreads();
    int p = it & 1;
    if (it + 1 < nIt) loadRegs((it + 1) * 32);
    v16h a = ldsA(Al[p] + w * 16 * 32, 32);
#pragma unroll
    for (int t = 0; t < 4; ++t) {
      v16h bt = ldsB_n(Wl[p] + t * 16 * 32, 32);
      acc[t] = wmma32(a, bt, acc[t]);
    }
    if (it + 1 < nIt) storeRegs((it + 1) & 1);
  }
  float scale = const_scale * (bscale ? bscale[b] : 1.0f);
  int n = lane & 15, mbase = (lane & 16) ? 8 : 0;
#pragma unroll
  for (int t = 0; t < 4; ++t) {
    int gn = n0 + t * 16 + n;
    int h = gn >> 6, dh = gn & 63;
#pragma unroll
    for (int r = 0; r < 8; ++r) {
      int gm = m0 + w * 16 + mbase + r;
      float bm = rowmul ? rowmul[(size_t)b * M + gm] : 1.0f;
      float val = (acc[t][r] + bias[gn] * bm) * scale;
      size_t idx = transposeOut
          ? ((((size_t)b * cH + h) * cDH + dh) * M + gm)
          : ((((size_t)b * cH + h) * M + gm) * cDH + dh);
      out[idx] = (_Float16)val;
    }
  }
}

// ---------------------------------------------------------------------------
// Attention: per (b,h), 128 query rows per workgroup, full K=256 in LDS.
// K/V/Q staged via TDM (tensor_load_to_lds, TENSORcnt) when available.
// scores = q@k^T (1/sqrt(DH) pre-folded into q), softmax over K, ctx=probs@v.
// grid (S/128, B*H), 256 threads (8 waves, 16 rows each). 144KB dynamic LDS.
// ---------------------------------------------------------------------------
__global__ __launch_bounds__(256)
void linformer_attn_kernel(const _Float16* __restrict__ Q,
                           const _Float16* __restrict__ Kf,
                           const _Float16* __restrict__ VfT,
                           float* __restrict__ out) {
  extern __shared__ char smem[];
  _Float16* kb = (_Float16*)smem;        // [key][dh] 256*64
  _Float16* vb = kb + cK * cDH;          // [dh][key] 64*256
  _Float16* qb = vb + cK * cDH;          // [row][dh] 128*64
  _Float16* pb = qb + 128 * cDH;         // 8 waves * [16][256] probs
  int tid = threadIdx.x, lane = tid & 31, w = tid >> 5;
  int bh = blockIdx.y;
  int b = bh / cH, h = bh % cH;
  int s0 = blockIdx.x * 128;
  const _Float16* kg = Kf + (size_t)bh * cK * cDH;
  const _Float16* vg = VfT + (size_t)bh * cK * cDH;
  const _Float16* qg = Q + ((size_t)bh * cS + s0) * cDH;
#if HAVE_TDM
  if (w == 0) {
    tdm_load_1d(kg, kb, cK * cDH / 4);    // 4096 x 8B
    tdm_load_1d(vg, vb, cK * cDH / 4);    // 4096 x 8B
    tdm_load_1d(qg, qb, 128 * cDH / 4);   // 2048 x 8B
    __builtin_amdgcn_s_wait_tensorcnt(0);
  }
#else
  for (int i = tid; i < cK * cDH / 8; i += 256) {
    ((uint4*)kb)[i] = ((const uint4*)kg)[i];
    ((uint4*)vb)[i] = ((const uint4*)vg)[i];
  }
  for (int i = tid; i < 128 * cDH / 8; i += 256)
    ((uint4*)qb)[i] = ((const uint4*)qg)[i];
#endif
  __syncthreads();

  int m0 = w * 16;
  v16h aq0 = ldsA(qb + m0 * cDH, cDH);
  v16h aq1 = ldsA(qb + m0 * cDH + 32, cDH);

  v8f sc[16];
#pragma unroll
  for (int kt = 0; kt < 16; ++kt) {
    v8f c = {};
    c = wmma32(aq0, ldsB_n(kb + kt * 16 * cDH, cDH), c);
    c = wmma32(aq1, ldsB_n(kb + kt * 16 * cDH + 32, cDH), c);
    sc[kt] = c;
  }

  // softmax over 256 cols: per-lane partial over the 16 tiles, then
  // xor-reduce across the 16 lanes of each half (masks 1..8 stay in-half).
  float inv[8];
#pragma unroll
  for (int r = 0; r < 8; ++r) {
    float m = sc[0][r];
#pragma unroll
    for (int kt = 1; kt < 16; ++kt) m = fmaxf(m, sc[kt][r]);
#pragma unroll
    for (int off = 1; off < 16; off <<= 1) m = fmaxf(m, __shfl_xor(m, off, 32));
    float s = 0.f;
#pragma unroll
    for (int kt = 0; kt < 16; ++kt) {
      float e = __expf(sc[kt][r] - m);
      sc[kt][r] = e;
      s += e;
    }
#pragma unroll
    for (int off = 1; off < 16; off <<= 1) s += __shfl_xor(s, off, 32);
    inv[r] = 1.0f / s;
  }

  // transpose probs to row-major f16 via per-wave private LDS region
  _Float16* pw = pb + (size_t)w * 16 * cK;
  {
    int n = lane & 15, mb = (lane & 16) ? 8 : 0;
#pragma unroll
    for (int kt = 0; kt < 16; ++kt)
#pragma unroll
      for (int r = 0; r < 8; ++r)
        pw[(mb + r) * cK + kt * 16 + n] = (_Float16)(sc[kt][r] * inv[r]);
  }

  // ctx = probs(16x256) @ v(256x64); A frag hoisted across the 4 n-tiles
  v8f cacc[4] = {};
#pragma unroll
  for (int kc = 0; kc < 8; ++kc) {
    v16h ap = ldsA(pw + kc * 32, cK);
#pragma unroll
    for (int nt = 0; nt < 4; ++nt)
      cacc[nt] = wmma32(ap, ldsB_n(vb + nt * 16 * cK + kc * 32, cK), cacc[nt]);
  }
  {
    int n = lane & 15, mb = (lane & 16) ? 8 : 0;
#pragma unroll
    for (int nt = 0; nt < 4; ++nt)
#pragma unroll
      for (int r = 0; r < 8; ++r) {
        int s = s0 + m0 + mb + r;
        int d = h * cDH + nt * 16 + n;
        out[((size_t)b * cS + s) * cD + d] = cacc[nt][r];
      }
  }
}

// ---------------------------------------------------------------------------
extern "C" void kernel_launch(void* const* d_in, const int* in_sizes, int n_in,
                              void* d_out, int out_size, void* d_ws, size_t ws_size,
                              hipStream_t stream) {
  (void)in_sizes; (void)n_in; (void)out_size; (void)ws_size;
  const float* hidden = (const float*)d_in[0];
  const float* mask   = (const float*)d_in[1];
  const float* Wq     = (const float*)d_in[2];
  const float* bq     = (const float*)d_in[3];
  const float* Wk     = (const float*)d_in[4];
  const float* bk     = (const float*)d_in[5];
  const float* Wv     = (const float*)d_in[6];
  const float* bv     = (const float*)d_in[7];
  const float* pk     = (const float*)d_in[8];
  const float* pv     = (const float*)d_in[9];
  float* out = (float*)d_out;

  char* ws = (char*)d_ws;
  float* invsqrt = (float*)ws;    ws += 256;
  float* csum    = (float*)ws;    ws += (size_t)2 * cB * cK * 4;
  float* Ck      = (float*)ws;    ws += (size_t)cB * cK * cD * 4;
  float* Cv      = (float*)ws;    ws += (size_t)cB * cK * cD * 4;
  _Float16* qf   = (_Float16*)ws; ws += (size_t)cB * cH * cS * cDH * 2;
  _Float16* kf   = (_Float16*)ws; ws += (size_t)cB * cH * cK * cDH * 2;
  _Float16* vfT  = (_Float16*)ws; ws += (size_t)cB * cH * cK * cDH * 2;
  _Float16* Xt   = (_Float16*)ws; ws += (size_t)cB * cD * cS * 2;

  len_kernel<<<cB, 256, 0, stream>>>(mask, invsqrt);
  csum_kernel<<<dim3(2, cB), 256, 0, stream>>>(mask, pk, pv, csum);
  transpose_x_kernel<<<dim3(cS / 64, cD / 64, cB), 256, 0, stream>>>(hidden, Xt);

  proj_pT_x_kernel<<<dim3(cD / 64, cK / 64, cB), 256, 0, stream>>>(
      Xt, mask, pk, pv, Ck, Cv);

  // q = (X Wq^T + bq) * (1/sqrt(DH)); k/v = (C W^T + csum*b) * invsqrt(len)
  gemm_xwT_kernel<<<dim3(cD / 64, cS / 128, cB), 256, 0, stream>>>(
      hidden, Wq, bq, nullptr, nullptr, 0.125f, qf, cS, 0);
  gemm_xwT_kernel<<<dim3(cD / 64, cK / 128, cB), 256, 0, stream>>>(
      Ck, Wk, bk, csum, invsqrt, 1.0f, kf, cK, 0);
  gemm_xwT_kernel<<<dim3(cD / 64, cK / 128, cB), 256, 0, stream>>>(
      Cv, Wv, bv, csum + (size_t)cB * cK, invsqrt, 1.0f, vfT, cK, 1);

  size_t attn_lds = (size_t)(cK * cDH + cK * cDH + 128 * cDH + 8 * 16 * cK) * sizeof(_Float16);
  linformer_attn_kernel<<<dim3(cS / 128, cB * cH), 256, attn_lds, stream>>>(
      qf, kf, vfT, out);
}
